// RitsModel_62431644615036
// MI455X (gfx1250) — compile-verified, matrix operand order checked by reading, not run on hardware
//
#include <hip/hip_runtime.h>
#include <hip/hip_bf16.h>

// ---------------------------------------------------------------------------
// RITS recurrent imputation on gfx1250 (MI455X), wave32 + WMMA f16->f32.
//   B=4096, T=48, D=64, H=128.
//   TWO waves cooperate on one 16-row batch tile. h lives as f16 in LDS;
//   c stays in f32 C-fragment registers. WMMA chains start from inline-zero
//   C; biases are loaded in-loop (loop-variant via runtime-zero stride) and
//   added in the f32 epilogue -> no hoisted vectors, no spills.
// ---------------------------------------------------------------------------

#define BB 4096
#define TT 48
#define DD 64
#define HH 128
#define RS (TT * DD)          // row stride in floats between batch rows
#define IMPUTE_WEIGHT 0.3f

typedef __attribute__((ext_vector_type(16))) _Float16 v16h;
typedef __attribute__((ext_vector_type(8)))  float    v8f;

// Fragment table offsets (units: fragments of 512 halves = 1KB each)
#define FRAG_TDH   0     // 2 kchunks x 8 ntiles
#define FRAG_HIST  16    // 4 x 4
#define FRAG_FEAT  32    // 2 x 4 (diag zeroed)
#define FRAG_COMB  40    // 4 x 4
#define FRAG_IH    56    // 4 x 32
#define FRAG_HH    184   // 4 x 32
#define NFRAGS     312
#define DENOM_OFF  (320 * 1024)   // byte offset of denom[] in workspace

// ---- fragment index helpers (16-bit A/B layout per CDNA5 ISA 7.12.2) ------
__device__ __forceinline__ int frag_k(int lane, int j) {
  int p = j >> 1, w = j & 1;
  return ((lane >> 4) << 3) + ((p & 4) << 2) + ((p & 3) << 1) + w;
}

__device__ __forceinline__ v8f splat8(float v) {
  v8f r;
#pragma unroll
  for (int i = 0; i < 8; ++i) r[i] = v;
  return r;
}

__device__ __forceinline__ v8f wmma_f16(v16h a, v16h b, v8f c) {
  return __builtin_amdgcn_wmma_f32_16x16x32_f16(false, a, false, b,
                                                (short)0, c, false, false);
}

// A-fragment (16x32 f16) gathered from row-major f32 global [16 rows, stride]
__device__ __forceinline__ v16h aglobal(const float* p, int stride, int kOff, int lane) {
  v16h a;
  const float* r = p + (size_t)(lane & 15) * stride;
#pragma unroll
  for (int j = 0; j < 16; ++j) a[j] = (_Float16)r[kOff + frag_k(lane, j)];
  return a;
}

// A-fragment of gamma_x = exp(-relu(d*diag(W_td_x)+b_td_x)), fused pointwise
__device__ __forceinline__ v16h agamma(const float* p, int stride, int kOff,
                                       const float* Wtdx, const float* btdx, int lane) {
  v16h a;
  const float* r = p + (size_t)(lane & 15) * stride;
#pragma unroll
  for (int j = 0; j < 16; ++j) {
    int k = kOff + frag_k(lane, j);
    float v = r[k] * Wtdx[k * 65] + btdx[k];
    v = v > 0.f ? v : 0.f;
    a[j] = (_Float16)__expf(-v);
  }
  return a;
}

// A-fragment from f16 LDS tile [16 x width]; consecutive half-pairs are
// adjacent in LDS, so load 8 x b32 instead of 16 x u16.
__device__ __forceinline__ v16h alds(const _Float16* lds, int width, int kOff, int lane) {
  v16h a;
  unsigned int* ap = (unsigned int*)&a;
  const unsigned int* r =
      (const unsigned int*)(lds + (lane & 15) * width + kOff);
  int kb = (lane >> 4) << 2;   // (lane>>4)*8 halves = *4 dwords
#pragma unroll
  for (int p = 0; p < 8; ++p)
    ap[p] = r[kb + ((p & 4) << 1) + (p & 3)];
  return a;
}

// C-fragment (16x16 f32) load/store: lane n = lane&15, rows mb..mb+7
__device__ __forceinline__ v8f cload(const float* p, int stride, int lane) {
  int n = lane & 15, mb = (lane >> 4) << 3;
  v8f c;
#pragma unroll
  for (int r = 0; r < 8; ++r) c[r] = p[(size_t)(mb + r) * stride + n];
  return c;
}
__device__ __forceinline__ void cstore(float* p, int stride, int lane, v8f c) {
  int n = lane & 15, mb = (lane >> 4) << 3;
#pragma unroll
  for (int r = 0; r < 8; ++r) p[(size_t)(mb + r) * stride + n] = c[r];
}
__device__ __forceinline__ void cstage(_Float16* lds, int width, int colBase, int lane, v8f c) {
  int n = lane & 15, mb = (lane >> 4) << 3;
#pragma unroll
  for (int r = 0; r < 8; ++r) lds[(mb + r) * width + colBase + n] = (_Float16)c[r];
}

__device__ __forceinline__ float sigmf(float x) { return 1.f / (1.f + __expf(-x)); }

// ---------------------------------------------------------------------------
// Kernel 1: pack weights (f32, row-major [rows, ld]) into f16 B-fragments.
// out[b][h] = sum_k in[b][k] * W[h][k]  =>  Bfrag[k][n] = W[ntile*16+n][ck*32+k]
// ---------------------------------------------------------------------------
__global__ __launch_bounds__(32) void pack_weights(
    const float* __restrict__ Wtdh, const float* __restrict__ Whist,
    const float* __restrict__ Wfeat, const float* __restrict__ Wcomb,
    const float* __restrict__ Wih, const float* __restrict__ Whh,
    _Float16* __restrict__ fragbuf) {
  int f = blockIdx.x, lane = threadIdx.x;
  const float* W; int ld, ck, nt; bool zerodiag = false;
  if (f < 16)       { W = Wtdh;  ld = 64;  int l = f;       ck = l >> 3; nt = l & 7; }
  else if (f < 32)  { W = Whist; ld = 128; int l = f - 16;  ck = l >> 2; nt = l & 3; }
  else if (f < 40)  { W = Wfeat; ld = 64;  int l = f - 32;  ck = l >> 2; nt = l & 3; zerodiag = true; }
  else if (f < 56)  { W = Wcomb; ld = 128; int l = f - 40;  ck = l >> 2; nt = l & 3; }
  else if (f < 184) { W = Wih;   ld = 128; int l = f - 56;  ck = l >> 5; nt = l & 31; }
  else              { W = Whh;   ld = 128; int l = f - 184; ck = l >> 5; nt = l & 31; }
  int n = lane & 15;
  _Float16* dst = fragbuf + (size_t)f * 512 + lane * 16;
#pragma unroll
  for (int j = 0; j < 16; ++j) {
    int k  = ck * 32 + frag_k(lane, j);
    int gn = nt * 16 + n;
    float v = W[(size_t)gn * ld + k];
    if (zerodiag && gn == k) v = 0.f;
    dst[j] = (_Float16)v;
  }
}

// ---------------------------------------------------------------------------
// Kernel 2: denom[t] = sum(masks[:,t,:]) + 1e-5  (pure-input reduction);
// block for t==0 also zeroes the loss scalar.
// ---------------------------------------------------------------------------
__global__ __launch_bounds__(256) void denom_kernel(
    const float* __restrict__ masks, float* __restrict__ denom, float* __restrict__ loss) {
  int t = blockIdx.x;
  float s = 0.f;
  for (int i = threadIdx.x; i < BB * DD; i += 256) {
    int b = i >> 6, d = i & 63;
    s += masks[(size_t)b * RS + (size_t)t * DD + d];
  }
  __shared__ float red[256];
  red[threadIdx.x] = s;
  __syncthreads();
  for (int off = 128; off; off >>= 1) {
    if (threadIdx.x < off) red[threadIdx.x] += red[threadIdx.x + off];
    __syncthreads();
  }
  if (threadIdx.x == 0) {
    denom[t] = red[0] + 1e-5f;
    if (t == 0) *loss = 0.f;
  }
}

// ---------------------------------------------------------------------------
// Kernel 3: main recurrent kernel. 128 blocks x 128 threads (4 waves = 2
// tiles, 2 cooperating waves per tile). Wave `half` owns H-tiles
// [half*4, half*4+4) and D-tiles [half*2, half*2+2).
// Per-timestep barrier schedule (all cross-wave LDS deps ordered):
//   [gamma: RMW own ldsH] B1 [read full aH; x_h; stage own x_c] B2
//   [read full aX] B3 [z_h; alpha; stage own c_c] B4
//   [read full aC; gates; write own h_new to ldsH]  -> next t
// ---------------------------------------------------------------------------
__global__ __launch_bounds__(128)
__attribute__((amdgpu_waves_per_eu(4)))
void rits_main(
    const float* __restrict__ values, const float* __restrict__ masks,
    const float* __restrict__ deltas,
    const float* __restrict__ b_td_h, const float* __restrict__ W_td_x,
    const float* __restrict__ b_td_x, const float* __restrict__ b_hist,
    const float* __restrict__ b_feat, const float* __restrict__ b_comb,
    const float* __restrict__ b_ih, const float* __restrict__ b_hh,
    const _Float16* __restrict__ fragbuf, const float* __restrict__ denom,
    float* __restrict__ loss, float* __restrict__ imps, int fbStride) {
  __shared__ _Float16 sH[2][16 * HH];   // h master copy, f16 (per tile)
  __shared__ _Float16 sX[2][16 * DD];   // x_c / c_c staging, f16 (per tile)

  const int lane = threadIdx.x & 31;
  const int wave = threadIdx.x >> 5;
  const int half = wave & 1;              // which half of N-tiles this wave owns
  const int tloc = wave >> 1;             // tile within block (0..1)
  const int rowBase = (blockIdx.x * 2 + tloc) * 16;
  const size_t rbOff = (size_t)rowBase * RS;
  _Float16* ldsH = sH[tloc];
  _Float16* ldsX = sX[tloc];
  const v16h* FB = (const v16h*)fragbuf;
  const int n = lane & 15;
  const int mb = (lane >> 4) << 3;

  // h0 = 0: both waves of the tile zero their halves of ldsH
  for (int idx = lane + half * 32; idx < 16 * HH; idx += 64)
    ldsH[idx] = (_Float16)0.f;
  __syncthreads();

  v8f cC[4];                              // 4 of 8 H-tiles of c per wave (f32)
#pragma unroll
  for (int i = 0; i < 4; ++i) cC[i] = splat8(0.f);
  float lacc = 0.f;

  for (int t = 0; t < TT; ++t) {
    // fbStride == 0 at runtime: same addresses, but loop-variant to the
    // compiler so neither fragments nor biases get hoisted/spilled.
    const size_t fv = (size_t)fbStride * (size_t)t;
    const v16h*  FBt  = FB + fv;
    const float* bTdT = b_td_h + fv;
    const float* bHiT = b_hist + fv;
    const float* bFeT = b_feat + fv;
    const float* bCoT = b_comb + fv;
    const float* bIhT = b_ih + fv;
    const float* bHhT = b_hh + fv;
    const float* Xt = values + rbOff + (size_t)t * DD;
    const float* Mt = masks  + rbOff + (size_t)t * DD;
    const float* Dt = deltas + rbOff + (size_t)t * DD;
    const float invden = 1.0f / denom[t];

    if (t + 1 < TT) {  // prefetch next step's rows (32 lanes cover 16 rows x 256B)
      size_t po = (size_t)(lane & 15) * RS + DD + (size_t)(lane >> 4) * 32;
      __builtin_prefetch(Xt + po);
      __builtin_prefetch(Mt + po);
      __builtin_prefetch(Dt + po);
    }

    // ---- gamma_h = exp(-relu(d @ W_td_h^T + b));  h *= gamma_h (LDS RMW) --
    v16h aD0 = aglobal(Dt, RS, 0, lane);
    v16h aD1 = aglobal(Dt, RS, 32, lane);
#pragma unroll
    for (int i = 0; i < 4; ++i) {
      int nt = half * 4 + i;
      v8f acc = wmma_f16(aD0, FBt[(FRAG_TDH + 0 * 8 + nt) * 32 + lane], splat8(0.f));
      acc = wmma_f16(aD1, FBt[(FRAG_TDH + 1 * 8 + nt) * 32 + lane], acc);
      float bt = bTdT[nt * 16 + n];
#pragma unroll
      for (int r = 0; r < 8; ++r) {
        float v = acc[r] + bt; v = v > 0.f ? v : 0.f;
        int off = (mb + r) * HH + nt * 16 + n;
        float hv = (float)ldsH[off];
        ldsH[off] = (_Float16)(hv * __expf(-v));
      }
    }
    __syncthreads();                                    // B1: decayed h staged

    // ---- x_h = h @ W_hist^T + b_hist  (aH kept live through gates) -------
    v16h aH0 = alds(ldsH, HH, 0, lane);
    v16h aH1 = alds(ldsH, HH, 32, lane);
    v16h aH2 = alds(ldsH, HH, 64, lane);
    v16h aH3 = alds(ldsH, HH, 96, lane);
    v8f xh[2];
#pragma unroll
    for (int i = 0; i < 2; ++i) {
      int nt = half * 2 + i;
      v8f acc = wmma_f16(aH0, FBt[(FRAG_HIST + 0 * 4 + nt) * 32 + lane], splat8(0.f));
      acc = wmma_f16(aH1, FBt[(FRAG_HIST + 1 * 4 + nt) * 32 + lane], acc);
      acc = wmma_f16(aH2, FBt[(FRAG_HIST + 2 * 4 + nt) * 32 + lane], acc);
      acc = wmma_f16(aH3, FBt[(FRAG_HIST + 3 * 4 + nt) * 32 + lane], acc);
      float bh = bHiT[nt * 16 + n];
#pragma unroll
      for (int r = 0; r < 8; ++r) acc[r] += bh;
      xh[i] = acc;
    }

    // ---- load x, m; loss1; x_c; stage x_c ---------------------------------
    v8f xC[2], mC[2];
#pragma unroll
    for (int i = 0; i < 2; ++i) {
      int nt = half * 2 + i;
      xC[i] = cload(Xt + nt * 16, RS, lane);
      mC[i] = cload(Mt + nt * 16, RS, lane);
    }
    float lsum = 0.f;
#pragma unroll
    for (int i = 0; i < 2; ++i) {
      int nt = half * 2 + i;
      v8f xc;
#pragma unroll
      for (int r = 0; r < 8; ++r) {
        float xv = xC[i][r], mv = mC[i][r], hv = xh[i][r];
        lsum += fabsf(xv - hv) * mv;
        xc[r] = mv * xv + (1.f - mv) * hv;
      }
      cstage(ldsX, DD, nt * 16, lane, xc);
    }
    __syncthreads();                                    // B2: x_c staged
    v16h aX0 = alds(ldsX, DD, 0, lane);
    v16h aX1 = alds(ldsX, DD, 32, lane);
    __syncthreads();                                    // B3: x_c consumed

    // ---- z_h = x_c @ W_feat_m^T + b_feat; loss2 ---------------------------
    v8f zh[2];
#pragma unroll
    for (int i = 0; i < 2; ++i) {
      int nt = half * 2 + i;
      v8f acc = wmma_f16(aX0, FBt[(FRAG_FEAT + 0 * 4 + nt) * 32 + lane], splat8(0.f));
      acc = wmma_f16(aX1, FBt[(FRAG_FEAT + 1 * 4 + nt) * 32 + lane], acc);
      float bf = bFeT[nt * 16 + n];
#pragma unroll
      for (int r = 0; r < 8; ++r) acc[r] += bf;
      zh[i] = acc;
#pragma unroll
      for (int r = 0; r < 8; ++r)
        lsum += fabsf(xC[i][r] - zh[i][r]) * mC[i][r];
    }

    // ---- alpha = [gamma_x, m] @ W_comb^T + b; c_h, loss3, c_c -------------
    v16h aG0 = agamma(Dt, RS, 0, W_td_x, b_td_x, lane);
    v16h aG1 = agamma(Dt, RS, 32, W_td_x, b_td_x, lane);
    v16h aM0 = aglobal(Mt, RS, 0, lane);
    v16h aM1 = aglobal(Mt, RS, 32, lane);
#pragma unroll
    for (int i = 0; i < 2; ++i) {
      int nt = half * 2 + i;
      v8f acc = wmma_f16(aG0, FBt[(FRAG_COMB + 0 * 4 + nt) * 32 + lane], splat8(0.f));
      acc = wmma_f16(aG1, FBt[(FRAG_COMB + 1 * 4 + nt) * 32 + lane], acc);
      acc = wmma_f16(aM0, FBt[(FRAG_COMB + 2 * 4 + nt) * 32 + lane], acc);
      acc = wmma_f16(aM1, FBt[(FRAG_COMB + 3 * 4 + nt) * 32 + lane], acc);
      float bc = bCoT[nt * 16 + n];
      v8f cc;
#pragma unroll
      for (int r = 0; r < 8; ++r) {
        float al = acc[r] + bc;
        float ch = al * zh[i][r] + (1.f - al) * xh[i][r];
        float xv = xC[i][r], mv = mC[i][r];
        lsum += fabsf(xv - ch) * mv;
        cc[r] = mv * xv + (1.f - mv) * ch;
      }
      cstore(imps + rbOff + (size_t)t * DD + nt * 16, RS, lane, cc);
      cstage(ldsX, DD, nt * 16, lane, cc);
    }
    lacc += lsum * invden;
    __syncthreads();                                    // B4: c_c staged

    // ---- LSTM gates: [c_c, m] @ W_ih^T + h @ W_hh^T; per-H-tile update ----
    v16h aC0 = alds(ldsX, DD, 0, lane);
    v16h aC1 = alds(ldsX, DD, 32, lane);
#pragma unroll
    for (int i = 0; i < 4; ++i) {
      int nt = half * 4 + i;
      v8f g[4];
#pragma unroll
      for (int q = 0; q < 4; ++q) g[q] = splat8(0.f);
#pragma unroll
      for (int ck = 0; ck < 8; ++ck) {
        v16h a = (ck == 0) ? aC0 : (ck == 1) ? aC1
               : (ck == 2) ? aM0 : (ck == 3) ? aM1
               : (ck == 4) ? aH0 : (ck == 5) ? aH1
               : (ck == 6) ? aH2 : aH3;
#pragma unroll
        for (int q = 0; q < 4; ++q) {
          int fidx = (ck < 4) ? (FRAG_IH + ck * 32 + q * 8 + nt)
                              : (FRAG_HH + (ck - 4) * 32 + q * 8 + nt);
          g[q] = wmma_f16(a, FBt[fidx * 32 + lane], g[q]);
        }
      }
      float bg0 = bIhT[0 * 128 + nt * 16 + n] + bHhT[0 * 128 + nt * 16 + n];
      float bg1 = bIhT[1 * 128 + nt * 16 + n] + bHhT[1 * 128 + nt * 16 + n];
      float bg2 = bIhT[2 * 128 + nt * 16 + n] + bHhT[2 * 128 + nt * 16 + n];
      float bg3 = bIhT[3 * 128 + nt * 16 + n] + bHhT[3 * 128 + nt * 16 + n];
      v8f hOut;
#pragma unroll
      for (int r = 0; r < 8; ++r) {
        float ig = sigmf(g[0][r] + bg0);
        float fg = sigmf(g[1][r] + bg1);
        float gg = tanhf(g[2][r] + bg2);
        float og = sigmf(g[3][r] + bg3);
        float cn = fg * cC[i][r] + ig * gg;
        cC[i][r] = cn;
        hOut[r] = og * tanhf(cn);
      }
      cstage(ldsH, HH, nt * 16, lane, hOut);  // own tiles; read next t (gamma)
    }
  }

  // wave-reduce lane-local loss partials, one atomic per wave
#pragma unroll
  for (int off = 16; off > 0; off >>= 1) lacc += __shfl_xor(lacc, off, 32);
  if (lane == 0) atomicAdd(loss, lacc * IMPUTE_WEIGHT);
}

// ---------------------------------------------------------------------------
extern "C" void kernel_launch(void* const* d_in, const int* in_sizes, int n_in,
                              void* d_out, int out_size, void* d_ws, size_t ws_size,
                              hipStream_t stream) {
  const float* values = (const float*)d_in[0];
  const float* masks  = (const float*)d_in[1];
  const float* deltas = (const float*)d_in[2];
  const float* W_td_h = (const float*)d_in[3];
  const float* b_td_h = (const float*)d_in[4];
  const float* W_td_x = (const float*)d_in[5];
  const float* b_td_x = (const float*)d_in[6];
  const float* W_hist = (const float*)d_in[7];
  const float* b_hist = (const float*)d_in[8];
  const float* W_feat = (const float*)d_in[9];
  const float* b_feat = (const float*)d_in[10];
  const float* W_comb = (const float*)d_in[11];
  const float* b_comb = (const float*)d_in[12];
  const float* W_ih   = (const float*)d_in[13];
  const float* W_hh   = (const float*)d_in[14];
  const float* b_ih   = (const float*)d_in[15];
  const float* b_hh   = (const float*)d_in[16];

  float* out = (float*)d_out;        // out[0] = loss, out[1..] = imputations
  _Float16* fragbuf = (_Float16*)d_ws;
  float* denomArr = (float*)((char*)d_ws + DENOM_OFF);

  pack_weights<<<NFRAGS, 32, 0, stream>>>(W_td_h, W_hist, W_feat, W_comb,
                                          W_ih, W_hh, fragbuf);
  denom_kernel<<<TT, 256, 0, stream>>>(masks, denomArr, out);
  rits_main<<<BB / 32, 128, 0, stream>>>(values, masks, deltas,
                                         b_td_h, W_td_x, b_td_x, b_hist, b_feat,
                                         b_comb, b_ih, b_hh,
                                         fragbuf, denomArr, out, out + 1,
                                         /*fbStride=*/0);
}
